// TimeSformerLayer_30846455120611
// MI455X (gfx1250) — compile-verified, matrix operand order checked by reading, not run on hardware
//
#include <hip/hip_runtime.h>
#include <hip/hip_bf16.h>
#include <math.h>

typedef __bf16 bf16;
typedef __bf16 v16bf __attribute__((ext_vector_type(16)));
typedef float  v8f   __attribute__((ext_vector_type(8)));

struct U32x4 { unsigned int x, y, z, w; };
union FragU { U32x4 u[2]; v16bf v; };

// ---------------- problem constants ----------------
#define C_B    8
#define C_T    8
#define C_HW   196
#define C_NH   12
#define C_D    768
#define C_DFF  3072
#define C_NTOK 1569          // 1 + HW*T
#define C_MT   12544         // B*HW*T   temporal tokens
#define C_MS   12608         // B*T*197  spatial tokens
#define C_MH   12552         // B*NTOK   hidden tokens
#define C_NMAX 197

// gfx1250 async global->LDS copy (ASYNCcnt-tracked), 16B per lane.
__device__ __forceinline__ void async_b128(void* lds, const void* gptr)
{
    const unsigned loff = (unsigned)(uintptr_t)lds;   // low 32 bits = LDS offset
    asm volatile("global_load_async_to_lds_b128 %0, %1, off"
                 :: "v"(loff), "v"(gptr) : "memory");
}
__device__ __forceinline__ void wait_async0()
{
    asm volatile("s_wait_asynccnt 0x0" ::: "memory");
}

// ---------------- GEMM: C = A(bf16)[M,K] x Bt(bf16)[N,K]^T (+bias, gelu) -------------
// Block tile 128x128, 8 waves, wave tile 32x64 (2x4 WMMA 16x16x32 bf16).
// Double-buffered LDS staging via global_load_async_to_lds_b128.
#define MTILE 128
#define NTILE 128
#define KTILE 32
#define LPITCH 48   // bf16 elems per LDS row (96B, keeps b128 reads 16B aligned)

__global__ __launch_bounds__(256) void gemm_bf16_wmma(
    const bf16* __restrict__ A, const bf16* __restrict__ Bt,
    const float* __restrict__ bias,
    float* __restrict__ outF, bf16* __restrict__ outH,
    int M, int N, int K, int fuseGelu)
{
    __shared__ bf16 As[2][MTILE][LPITCH];
    __shared__ bf16 Bs[2][NTILE][LPITCH];   // Bt tile: Bs[n][k]

    const int tid  = threadIdx.x;
    const int lane = tid & 31;
    const int wave = tid >> 5;
    const int waveM = wave & 3;          // 4 waves along M (32 rows each)
    const int waveN = wave >> 2;         // 2 waves along N (64 cols each)
    const int lane16   = lane & 15;
    const int laneHalf = lane >> 4;
    const int kBase    = laneHalf * 8;   // fragment K base within a 32-K tile

    const int blockM = blockIdx.y * MTILE;
    const int blockN = blockIdx.x * NTILE;

    v8f acc[2][4];
    #pragma unroll
    for (int mi = 0; mi < 2; ++mi)
        #pragma unroll
        for (int ni = 0; ni < 4; ++ni)
            #pragma unroll
            for (int e = 0; e < 8; ++e) acc[mi][ni][e] = 0.0f;

    // stage one 128x32 A tile + 128x32 B tile into LDS buffer `buf`
    auto stage = [&](int buf, int k0) {
        #pragma unroll
        for (int i = 0; i < 2; ++i) {
            const int ch  = tid + i * 256;      // 512 chunks of 8 bf16 per matrix
            const int row = ch >> 2;
            const int seg = (ch & 3) * 8;
            int gr = blockM + row;
            gr = (gr < M) ? gr : (M - 1);       // clamp: OOB rows masked at store
            async_b128(&As[buf][row][seg], A  + (size_t)gr * K + k0 + seg);
            async_b128(&Bs[buf][row][seg], Bt + (size_t)(blockN + row) * K + k0 + seg);
        }
    };

    stage(0, 0);
    const int nk = K / KTILE;
    for (int it = 0; it < nk; ++it) {
        const int buf = it & 1;
        wait_async0();                      // my deposits for tile `it` done
        __syncthreads();                    // everyone's deposits visible
        if (it + 1 < nk) stage(buf ^ 1, (it + 1) * KTILE);  // prefetch next tile

        FragU aF[2], bF[4];
        #pragma unroll
        for (int mi = 0; mi < 2; ++mi) {
            const int row = waveM * 32 + mi * 16 + lane16;
            aF[mi].u[0] = *reinterpret_cast<const U32x4*>(&As[buf][row][kBase]);
            aF[mi].u[1] = *reinterpret_cast<const U32x4*>(&As[buf][row][16 + kBase]);
        }
        #pragma unroll
        for (int ni = 0; ni < 4; ++ni) {
            const int col = waveN * 64 + ni * 16 + lane16;
            bF[ni].u[0] = *reinterpret_cast<const U32x4*>(&Bs[buf][col][kBase]);
            bF[ni].u[1] = *reinterpret_cast<const U32x4*>(&Bs[buf][col][16 + kBase]);
        }
        #pragma unroll
        for (int mi = 0; mi < 2; ++mi)
            #pragma unroll
            for (int ni = 0; ni < 4; ++ni)
                acc[mi][ni] = __builtin_amdgcn_wmma_f32_16x16x32_bf16(
                    false, aF[mi].v, false, bF[ni].v,
                    (short)0, acc[mi][ni], false, false);
        __syncthreads();
    }

    // --- epilogue: C layout (lane16 = N col, VGPR r -> M = r + 8*laneHalf)
    #pragma unroll
    for (int ni = 0; ni < 4; ++ni) {
        const int col = blockN + waveN * 64 + ni * 16 + lane16;
        const float bv = bias ? bias[col] : 0.0f;
        #pragma unroll
        for (int mi = 0; mi < 2; ++mi) {
            #pragma unroll
            for (int r = 0; r < 8; ++r) {
                const int rowg = blockM + waveM * 32 + mi * 16 + laneHalf * 8 + r;
                if (rowg < M) {
                    float v = acc[mi][ni][r] + bv;
                    if (fuseGelu) v = 0.5f * v * (1.0f + erff(v * 0.70710678118f));
                    if (outF) outF[(size_t)rowg * N + col] = v;
                    if (outH) outH[(size_t)rowg * N + col] = (bf16)v;
                }
            }
        }
    }
}

// ---------------- LayerNorm (fused gather) -> bf16 ----------------
// mode 0: rows of srcA          (srcA = [M, 768] f32)
// mode 1: hidden skip-CLS view  (srcA = hidden,  row r -> hidden[b,1+j])
// mode 2: spatial gather        (srcA = hidden for CLS, srcB = xt buffer)
__global__ __launch_bounds__(256) void ln_bf16_kernel(
    const float* __restrict__ srcA, const float* __restrict__ srcB,
    const float* __restrict__ g, const float* __restrict__ bta,
    bf16* __restrict__ out, int M, int mode)
{
    const int r = blockIdx.x;
    if (r >= M) return;
    const float* src;
    if (mode == 0) {
        src = srcA + (size_t)r * C_D;
    } else if (mode == 1) {
        const int b = r / 1568, j = r % 1568;
        src = srcA + ((size_t)b * C_NTOK + 1 + j) * C_D;
    } else {
        const int s = r / 197, i = r % 197;
        const int b = s / C_T, t = s % C_T;
        if (i == 0) src = srcA + (size_t)b * C_NTOK * C_D;           // CLS
        else        src = srcB + ((size_t)b * 1568 + (i - 1) * C_T + t) * C_D;
    }

    __shared__ float sred[256];
    const int tid = threadIdx.x;
    const float x0 = src[tid], x1 = src[tid + 256], x2 = src[tid + 512];
    sred[tid] = x0 + x1 + x2;
    __syncthreads();
    #pragma unroll
    for (int off = 128; off > 0; off >>= 1) {
        if (tid < off) sred[tid] += sred[tid + off];
        __syncthreads();
    }
    const float mean = sred[0] * (1.0f / C_D);
    __syncthreads();
    const float d0 = x0 - mean, d1 = x1 - mean, d2 = x2 - mean;
    sred[tid] = d0 * d0 + d1 * d1 + d2 * d2;
    __syncthreads();
    #pragma unroll
    for (int off = 128; off > 0; off >>= 1) {
        if (tid < off) sred[tid] += sred[tid + off];
        __syncthreads();
    }
    const float rinv = rsqrtf(sred[0] * (1.0f / C_D) + 1e-6f);
    bf16* o = out + (size_t)r * C_D;
    o[tid]       = (bf16)(d0 * rinv * g[tid]       + bta[tid]);
    o[tid + 256] = (bf16)(d1 * rinv * g[tid + 256] + bta[tid + 256]);
    o[tid + 512] = (bf16)(d2 * rinv * g[tid + 512] + bta[tid + 512]);
}

// ---------------- attention: one block per (sequence, head) ----------------
// qkv: [S*N, 2304] bf16 (cols = 3 x NH x 64). ctx: [S*N, 768] bf16.
__global__ __launch_bounds__(256) void attention_kernel(
    const bf16* __restrict__ qkv, bf16* __restrict__ ctx,
    int S, int N, float scale)
{
    const int s = blockIdx.x / C_NH;
    const int h = blockIdx.x % C_NH;
    __shared__ bf16 ks[C_NMAX * 64];
    __shared__ bf16 vs[C_NMAX * 64];
    __shared__ float probs[256];
    __shared__ float qrow[64];
    __shared__ float sred[256];

    const int tid = threadIdx.x;
    const size_t rowBase = (size_t)s * N * 2304;
    const int hc = h * 64;

    for (int idx = tid; idx < N * 64; idx += 256) {
        const int m = idx >> 6, d = idx & 63;
        ks[idx] = qkv[rowBase + (size_t)m * 2304 +  768 + hc + d];
        vs[idx] = qkv[rowBase + (size_t)m * 2304 + 1536 + hc + d];
    }
    __syncthreads();

    for (int n = 0; n < N; ++n) {
        if (tid < 64) qrow[tid] = (float)qkv[rowBase + (size_t)n * 2304 + hc + tid];
        __syncthreads();

        float sc = -3.0e38f;
        if (tid < N) {
            float a = 0.0f;
            #pragma unroll 8
            for (int d = 0; d < 64; ++d) a += qrow[d] * (float)ks[tid * 64 + d];
            sc = a * scale;
        }
        sred[tid] = sc;
        __syncthreads();
        #pragma unroll
        for (int off = 128; off > 0; off >>= 1) {
            if (tid < off) sred[tid] = fmaxf(sred[tid], sred[tid + off]);
            __syncthreads();
        }
        const float maxv = sred[0];
        __syncthreads();

        const float p = (tid < N) ? __expf(sc - maxv) : 0.0f;
        probs[tid] = p;
        sred[tid]  = p;
        __syncthreads();
        #pragma unroll
        for (int off = 128; off > 0; off >>= 1) {
            if (tid < off) sred[tid] += sred[tid + off];
            __syncthreads();
        }
        const float inv = 1.0f / sred[0];

        if (tid < 64) {
            float a = 0.0f;
            for (int m = 0; m < N; ++m) a += probs[m] * (float)vs[m * 64 + tid];
            ctx[((size_t)s * N + n) * C_D + hc + tid] = (bf16)(a * inv);
        }
        __syncthreads();
    }
}

// ---------------- elementwise helpers ----------------
// W[K,N] f32 -> Wt[N,K] bf16 (transposed weight cast; done once, tiny vs GEMMs)
__global__ __launch_bounds__(256) void transpose_cast_kernel(
    const float* __restrict__ W, bf16* __restrict__ Wt, int Kd, int Nd)
{
    const int total = Kd * Nd;
    for (int idx = blockIdx.x * 256 + threadIdx.x; idx < total; idx += gridDim.x * 256) {
        const int k = idx / Nd, n = idx % Nd;
        Wt[(size_t)n * Kd + k] = (bf16)W[idx];
    }
}

// xt = hidden[:,1:,:] + denseOut
__global__ __launch_bounds__(256) void residual_xt_kernel(
    const float* __restrict__ hidden, const float* __restrict__ denseOut,
    float* __restrict__ xt, int total)
{
    for (int idx = blockIdx.x * 256 + threadIdx.x; idx < total; idx += gridDim.x * 256) {
        const int c = idx % C_D, m = idx / C_D;
        const int b = m / 1568, j = m % 1568;
        xt[idx] = hidden[((size_t)b * C_NTOK + 1 + j) * C_D + c] + denseOut[idx];
    }
}

// hs = concat(cls, xt) + concat(mean_t cls_ctx, res_s permuted)
__global__ __launch_bounds__(256) void build_hs_kernel(
    const float* __restrict__ hidden, const float* __restrict__ xt,
    const float* __restrict__ sOut, float* __restrict__ hs, int total)
{
    for (int idx = blockIdx.x * 256 + threadIdx.x; idx < total; idx += gridDim.x * 256) {
        const int c = idx % C_D;
        const int row = idx / C_D;
        const int b = row / C_NTOK, i = row % C_NTOK;
        float base, res;
        if (i == 0) {
            base = hidden[idx];
            float a = 0.0f;
            #pragma unroll
            for (int t = 0; t < C_T; ++t)
                a += sOut[((size_t)(b * C_T + t) * 197) * C_D + c];
            res = a * 0.125f;
        } else {
            const int j = i - 1, p = j >> 3, t = j & 7;
            base = xt[((size_t)b * 1568 + j) * C_D + c];
            res  = sOut[((size_t)(b * C_T + t) * 197 + 1 + p) * C_D + c];
        }
        hs[idx] = base + res;
    }
}

__global__ __launch_bounds__(256) void add_kernel(
    const float* __restrict__ a, const float* __restrict__ b,
    float* __restrict__ o, int n)
{
    for (int i = blockIdx.x * 256 + threadIdx.x; i < n; i += gridDim.x * 256)
        o[i] = a[i] + b[i];
}

// ---------------- host side ----------------
static inline void launch_gemm(const bf16* A, const bf16* Bt, const float* bias,
                               float* outF, bf16* outH, int M, int N, int K,
                               int gelu, hipStream_t st)
{
    dim3 grid(N / NTILE, (M + MTILE - 1) / MTILE);
    gemm_bf16_wmma<<<grid, 256, 0, st>>>(A, Bt, bias, outF, outH, M, N, K, gelu);
}

extern "C" void kernel_launch(void* const* d_in, const int* in_sizes, int n_in,
                              void* d_out, int out_size, void* d_ws, size_t ws_size,
                              hipStream_t stream)
{
    (void)in_sizes; (void)n_in; (void)out_size; (void)ws_size;
    const float* hidden  = (const float*)d_in[0];
    const float* t_ln_g  = (const float*)d_in[1];
    const float* t_ln_b  = (const float*)d_in[2];
    const float* t_qkv_w = (const float*)d_in[3];
    const float* t_qkv_b = (const float*)d_in[4];
    const float* t_out_w = (const float*)d_in[5];
    const float* t_out_b = (const float*)d_in[6];
    const float* t_dns_w = (const float*)d_in[7];
    const float* t_dns_b = (const float*)d_in[8];
    const float* ln1_g   = (const float*)d_in[9];
    const float* ln1_b   = (const float*)d_in[10];
    const float* s_qkv_w = (const float*)d_in[11];
    const float* s_qkv_b = (const float*)d_in[12];
    const float* s_out_w = (const float*)d_in[13];
    const float* s_out_b = (const float*)d_in[14];
    const float* ln2_g   = (const float*)d_in[15];
    const float* ln2_b   = (const float*)d_in[16];
    const float* fc1_w   = (const float*)d_in[17];
    const float* fc1_b   = (const float*)d_in[18];
    const float* fc2_w   = (const float*)d_in[19];
    const float* fc2_b   = (const float*)d_in[20];
    float* out = (float*)d_out;

    // --- workspace carve (~390 MB total) ---
    char* base = (char*)d_ws;
    size_t off = 0;
    auto carve = [&](size_t bytes) -> void* {
        void* p = base + off;
        off = (off + bytes + 255) & ~(size_t)255;
        return p;
    };
    bf16* wQKVt = (bf16*)carve((size_t)C_D * 3 * C_D * 2);   // transposed [N,K]
    bf16* wOutT = (bf16*)carve((size_t)C_D * C_D * 2);
    bf16* wDns  = (bf16*)carve((size_t)C_D * C_D * 2);
    bf16* wQKVs = (bf16*)carve((size_t)C_D * 3 * C_D * 2);
    bf16* wOutS = (bf16*)carve((size_t)C_D * C_D * 2);
    bf16* wFc1  = (bf16*)carve((size_t)C_D * C_DFF * 2);
    bf16* wFc2  = (bf16*)carve((size_t)C_DFF * C_D * 2);
    bf16* lnBuf   = (bf16*)carve((size_t)C_MS * C_D * 2);
    bf16* qkvBuf  = (bf16*)carve((size_t)C_MS * 3 * C_D * 2);
    bf16* ctxBuf  = (bf16*)carve((size_t)C_MS * C_D * 2);
    bf16* geluBuf = (bf16*)carve((size_t)C_MH * C_DFF * 2);
    float* denseOut = (float*)carve((size_t)C_MT * C_D * 4);
    float* xtBuf    = (float*)carve((size_t)C_MT * C_D * 4);
    float* sOut     = (float*)carve((size_t)C_MS * C_D * 4);
    float* hsBuf    = (float*)carve((size_t)C_MH * C_D * 4);
    float* fc2Out   = (float*)carve((size_t)C_MH * C_D * 4);

    // --- transpose-cast weights to bf16 [N,K] ---
    transpose_cast_kernel<<<1024, 256, 0, stream>>>(t_qkv_w, wQKVt, C_D, 3 * C_D);
    transpose_cast_kernel<<<1024, 256, 0, stream>>>(t_out_w, wOutT, C_D, C_D);
    transpose_cast_kernel<<<1024, 256, 0, stream>>>(t_dns_w, wDns,  C_D, C_D);
    transpose_cast_kernel<<<1024, 256, 0, stream>>>(s_qkv_w, wQKVs, C_D, 3 * C_D);
    transpose_cast_kernel<<<1024, 256, 0, stream>>>(s_out_w, wOutS, C_D, C_D);
    transpose_cast_kernel<<<1024, 256, 0, stream>>>(fc1_w,   wFc1,  C_D, C_DFF);
    transpose_cast_kernel<<<1024, 256, 0, stream>>>(fc2_w,   wFc2,  C_DFF, C_D);

    const float scale = 0.125f;  // 64^-0.5

    // === temporal attention ===
    ln_bf16_kernel<<<C_MT, 256, 0, stream>>>(hidden, nullptr, t_ln_g, t_ln_b,
                                             lnBuf, C_MT, 1);
    launch_gemm(lnBuf, wQKVt, t_qkv_b, nullptr, qkvBuf, C_MT, 3 * C_D, C_D, 0, stream);
    attention_kernel<<<1568 * C_NH, 256, 0, stream>>>(qkvBuf, ctxBuf, 1568, C_T, scale);
    launch_gemm(ctxBuf, wOutT, t_out_b, nullptr, lnBuf, C_MT, C_D, C_D, 0, stream);  // reuse lnBuf
    launch_gemm(lnBuf, wDns, t_dns_b, denseOut, nullptr, C_MT, C_D, C_D, 0, stream);
    residual_xt_kernel<<<8192, 256, 0, stream>>>(hidden, denseOut, xtBuf, C_MT * C_D);

    // === spatial attention ===
    ln_bf16_kernel<<<C_MS, 256, 0, stream>>>(hidden, xtBuf, ln1_g, ln1_b,
                                             lnBuf, C_MS, 2);
    launch_gemm(lnBuf, wQKVs, s_qkv_b, nullptr, qkvBuf, C_MS, 3 * C_D, C_D, 0, stream);
    attention_kernel<<<64 * C_NH, 256, 0, stream>>>(qkvBuf, ctxBuf, 64, 197, scale);
    launch_gemm(ctxBuf, wOutS, s_out_b, sOut, nullptr, C_MS, C_D, C_D, 0, stream);
    build_hs_kernel<<<8192, 256, 0, stream>>>(hidden, xtBuf, sOut, hsBuf, C_MH * C_D);

    // === MLP ===
    ln_bf16_kernel<<<C_MH, 256, 0, stream>>>(hsBuf, nullptr, ln2_g, ln2_b,
                                             lnBuf, C_MH, 0);
    launch_gemm(lnBuf, wFc1, fc1_b, nullptr, geluBuf, C_MH, C_DFF, C_D, 1, stream);
    launch_gemm(geluBuf, wFc2, fc2_b, fc2Out, nullptr, C_MH, C_D, C_DFF, 0, stream);
    add_kernel<<<8192, 256, 0, stream>>>(hsBuf, fc2Out, out, C_MH * C_D);
}
